// ProductPhasorBlock_84825604096584
// MI455X (gfx1250) — matrix operand order repaired and checked by reading, not compiled, not run on hardware
//
#include <hip/hip_runtime.h>
#include <math.h>

// Problem constants (B, L, D, K from the reference)
#define B_ 4
#define L_ 1024
#define D_ 256
#define K_ 64
#define N_ (B_ * L_)
#define PI_F 3.14159265358979323846f

typedef __attribute__((ext_vector_type(16))) _Float16 v16h;
typedef __attribute__((ext_vector_type(8)))  _Float16 v8h;
typedef __attribute__((ext_vector_type(4)))  _Float16 v4h;
typedef __attribute__((ext_vector_type(8)))  float    v8f;
typedef __attribute__((ext_vector_type(4)))  float    v4f;

// ---------------------------------------------------------------------------
// CDNA5 async global->LDS copy (16B per lane) + wait
// ---------------------------------------------------------------------------
__device__ __forceinline__ unsigned lds_addr(const void* p) {
  // generic pointer to LDS: low 32 bits are the LDS byte address
  return (unsigned)(unsigned long long)p;
}
__device__ __forceinline__ void async_b128(unsigned lds, const void* g) {
  asm volatile("global_load_async_to_lds_b128 %0, %1, off"
               :: "v"(lds), "v"(g) : "memory");
}
__device__ __forceinline__ void wait_async0() {
  asm volatile("s_wait_asynccnt 0x0" ::: "memory");
}

// ---------------------------------------------------------------------------
// WMMA helper: D = A(16x32 f16) * B(32x16 f16) + C(16x16 f32)
// ---------------------------------------------------------------------------
__device__ __forceinline__ v8f wmma_f16(v16h a, v16h b, v8f c) {
  return __builtin_amdgcn_wmma_f32_16x16x32_f16(
      false, a, false, b, (short)0, c, false, false);
}

// A-operand fragment (16x32 f16) from row-major M[m][k]; m = lane%16.
// Per-lane data = two contiguous 8-half runs: k = 8g+0..7 and 16+8g+0..7.
__device__ __forceinline__ v16h load_fragA(const _Float16* base, int ld) {
  const int lane = threadIdx.x & 31;
  const int g = lane >> 4, r = lane & 15;
  const _Float16* p = base + r * ld + 8 * g;
  v8h lo = *(const v8h*)p;         // k = 8g .. 8g+7
  v8h hi = *(const v8h*)(p + 16);  // k = 16+8g .. 16+8g+7
  return __builtin_shufflevector(lo, hi, 0, 1, 2, 3, 4, 5, 6, 7,
                                 8, 9, 10, 11, 12, 13, 14, 15);
}

// B-operand fragment (32x16 f16) where memory holds M^T row-major: M[n][k].
// ISA B layout: lane n = lane%16, k = 8g+0..7 and 16+8g+0..7 (matches fragA).
__device__ __forceinline__ v16h load_fragBt(const _Float16* base, int ld) {
  const int lane = threadIdx.x & 31;
  const int g = lane >> 4, r = lane & 15;
  const _Float16* p = base + r * ld + 8 * g;
  v8h lo = *(const v8h*)p;
  v8h hi = *(const v8h*)(p + 16);
  return __builtin_shufflevector(lo, hi, 0, 1, 2, 3, 4, 5, 6, 7,
                                 8, 9, 10, 11, 12, 13, 14, 15);
}

// ---------------------------------------------------------------------------
// Kernel 0a: fp32 -> f16
// Kernel 0b: fp32 -> f16 with transpose (src R x C  ->  dst C x R)
// ---------------------------------------------------------------------------
__global__ void cvt_f32_to_f16(const float* __restrict__ src,
                               _Float16* __restrict__ dst, int n) {
  int i = blockIdx.x * 256 + threadIdx.x;
  if (i < n) dst[i] = (_Float16)src[i];
}
__global__ void cvt_transpose(const float* __restrict__ src,
                              _Float16* __restrict__ dst, int R, int C) {
  int j = blockIdx.x * 256 + threadIdx.x;  // dst index; dst[n*R + k]
  if (j < R * C) {
    int n = j / R, k = j % R;
    dst[j] = (_Float16)src[k * C + n];
  }
}

// ---------------------------------------------------------------------------
// Kernel 1: encoder. One workgroup (8 waves) per 16-row tile.
// Weights are pre-transposed: W*t[out][in]. V stored transposed: Vt[b][d][l].
// ---------------------------------------------------------------------------
__global__ __launch_bounds__(256) void encoder_kernel(
    const _Float16* __restrict__ xh,
    const _Float16* __restrict__ W1t, const float* __restrict__ b1,
    const _Float16* __restrict__ W2t, const float* __restrict__ b2,
    const _Float16* __restrict__ Wat, const float* __restrict__ ba,
    const _Float16* __restrict__ Wvt, const float* __restrict__ bv,
    const float* __restrict__ pos_phases,
    _Float16* __restrict__ PR, _Float16* __restrict__ PIM,
    _Float16* __restrict__ Vt) {
  __shared__ __attribute__((aligned(16))) _Float16 xt[16 * D_];   // 8 KB
  __shared__ __attribute__((aligned(16))) _Float16 Ht[16 * D_];   // 8 KB
  __shared__ __attribute__((aligned(16))) float    phs[16 * K_];  // 4 KB
  __shared__ __attribute__((aligned(16))) float    amps[16 * K_]; // 4 KB

  const int row0 = blockIdx.x * 16;
  const int tid  = threadIdx.x;
  const int wave = tid >> 5;
  const int lane = tid & 31;
  const int g = lane >> 4, r = lane & 15;

  // async stage x tile into LDS: 256 threads x 32B
  {
    const int m = tid >> 4, c0 = (tid & 15) * 16;
    const _Float16* gx = xh + (size_t)(row0 + m) * D_ + c0;
    const unsigned l0 = lds_addr(xt) + (unsigned)(m * D_ + c0) * 2u;
    async_b128(l0, gx);
    async_b128(l0 + 16u, gx + 8);
    wait_async0();
  }
  __syncthreads();

  // ---- Phase 1: H = GELU(x@W1 + b1) ----
  for (int ct = wave; ct < D_ / 16; ct += 8) {
    v8f acc = {};
#pragma unroll
    for (int ks = 0; ks < D_ / 32; ++ks) {
      v16h a = load_fragA(xt + ks * 32, D_);
      v16h b = load_fragBt(W1t + (ct * 16) * D_ + ks * 32, D_);
      acc = wmma_f16(a, b, acc);
    }
    const int n = ct * 16 + r;
    const float bias = b1[n];
#pragma unroll
    for (int j = 0; j < 8; ++j) {
      const int m = j + 8 * g;
      const float v = acc[j] + bias;
      Ht[m * D_ + n] = (_Float16)(0.5f * v *
                                  (1.0f + erff(v * 0.70710678118654752f)));
    }
  }
  __syncthreads();

  // ---- Phase 2: waves 0-3 -> phase (4 tiles of K); waves 4-7 -> amp ----
  if (wave < 4) {
    const int ct = wave;
    v8f acc = {};
#pragma unroll
    for (int ks = 0; ks < D_ / 32; ++ks) {
      v16h a = load_fragA(Ht + ks * 32, D_);
      v16h b = load_fragBt(W2t + (ct * 16) * D_ + ks * 32, D_);
      acc = wmma_f16(a, b, acc);
    }
    const int n = ct * 16 + r;
    const float bias = b2[n];
#pragma unroll
    for (int j = 0; j < 8; ++j) {
      const int m = j + 8 * g;
      const int l = (row0 + m) & (L_ - 1);
      phs[m * K_ + n] = pos_phases[l * K_ + n] + tanhf(acc[j] + bias) * PI_F;
    }
  } else {
    const int ct = wave - 4;
    v8f acc = {};
#pragma unroll
    for (int ks = 0; ks < D_ / 32; ++ks) {
      v16h a = load_fragA(xt + ks * 32, D_);
      v16h b = load_fragBt(Wat + (ct * 16) * D_ + ks * 32, D_);
      acc = wmma_f16(a, b, acc);
    }
    const int n = ct * 16 + r;
    const float bias = ba[n];
#pragma unroll
    for (int j = 0; j < 8; ++j) {
      const int m = j + 8 * g;
      const float v = acc[j] + bias;
      const float sp = (v > 20.0f) ? v : log1pf(expf(v));
      amps[m * K_ + n] = sp + 0.1f;
    }
  }
  __syncthreads();

  // ---- elementwise pr/pim, vectorized b64 stores ----
  {
    const int e0 = tid * 4;  // 16*K_ = 1024 = 256 threads * 4
    const int m = e0 >> 6, k = e0 & (K_ - 1);
    v4h prv, piv;
#pragma unroll
    for (int u = 0; u < 4; ++u) {
      const float ph = phs[e0 + u], am = amps[e0 + u];
      prv[u] = (_Float16)(am * cosf(ph));
      piv[u] = (_Float16)(am * sinf(ph));
    }
    *(v4h*)(PR + (size_t)(row0 + m) * K_ + k)  = prv;
    *(v4h*)(PIM + (size_t)(row0 + m) * K_ + k) = piv;
  }

  // ---- Phase 3: V = x@Wv + bv, stored TRANSPOSED Vt[b][d][l] ----
  const int bidx = row0 / L_;
  const int l0 = row0 & (L_ - 1);
  for (int ct = wave; ct < D_ / 16; ct += 8) {
    v8f acc = {};
#pragma unroll
    for (int ks = 0; ks < D_ / 32; ++ks) {
      v16h a = load_fragA(xt + ks * 32, D_);
      v16h b = load_fragBt(Wvt + (ct * 16) * D_ + ks * 32, D_);
      acc = wmma_f16(a, b, acc);
    }
    const int d = ct * 16 + r;
    const float bias = bv[d];
    v8h vv;
#pragma unroll
    for (int j = 0; j < 8; ++j) vv[j] = (_Float16)(acc[j] + bias);
    // rows l0+8g .. l0+8g+7 are contiguous -> one 16B store
    *(v8h*)(Vt + ((size_t)bidx * D_ + d) * L_ + l0 + 8 * g) = vv;
  }
}

// ---------------------------------------------------------------------------
// Kernel 2: causal linear attention.
// One workgroup = 4 waves = 4 consecutive query tiles of one batch.
// Key pair-blocks (32 keys) async-staged in LDS, shared by all 4 waves.
// S^T = PRk@PRq^T + PIMk@PIMq^T : C/D layout == A-operand layout of S.
// ---------------------------------------------------------------------------
__global__ __launch_bounds__(128) void attn_kernel(
    const _Float16* __restrict__ PR, const _Float16* __restrict__ PIM,
    const _Float16* __restrict__ Vt, float* __restrict__ RETt) {
  __shared__ __attribute__((aligned(16))) _Float16 PRs[32 * K_];  // 4 KB
  __shared__ __attribute__((aligned(16))) _Float16 PIMs[32 * K_]; // 4 KB
  __shared__ __attribute__((aligned(16))) _Float16 Vts[D_ * 32];  // 16 KB [d][key]

  const int b   = blockIdx.x >> 4;  // 16 query groups per batch
  const int qt0 = (blockIdx.x & 15) * 4;
  const int tid = threadIdx.x;
  const int wave = tid >> 5;
  const int lane = tid & 31;
  const int g = lane >> 4, r = lane & 15;
  const int qt = qt0 + wave;
  const int rb = b * L_ + qt * 16;

  // query B-operand fragments (two 32-wide k-steps), constant over key loop
  const v16h prq0 = load_fragBt(PR + (size_t)rb * K_, K_);
  const v16h prq1 = load_fragBt(PR + (size_t)rb * K_ + 32, K_);
  const v16h piq0 = load_fragBt(PIM + (size_t)rb * K_, K_);
  const v16h piq1 = load_fragBt(PIM + (size_t)rb * K_ + 32, K_);

  v8f acc[D_ / 16] = {};

  const int ktmax = qt0 + 3;
  for (int kt2 = 0; kt2 <= ktmax; kt2 += 2) {
    const int krb = b * L_ + kt2 * 16;
    // ---- async stage PR/PIM (4KB each) + Vt block (16KB) into LDS ----
    {
      const _Float16* gpr = PR + (size_t)krb * K_;
      const _Float16* gpi = PIM + (size_t)krb * K_;
#pragma unroll
      for (int i = 0; i < 2; ++i) {
        const int c = i * 128 + tid;  // 256 chunks of 16B
        async_b128(lds_addr(PRs) + (unsigned)c * 16u, gpr + c * 8);
        async_b128(lds_addr(PIMs) + (unsigned)c * 16u, gpi + c * 8);
      }
      const _Float16* gvt = Vt + (size_t)b * D_ * L_ + kt2 * 16;
#pragma unroll
      for (int i = 0; i < 8; ++i) {
        const int c = i * 128 + tid;  // 1024 chunks of 16B
        const int d = c >> 2;         // 4 chunks per 32-key row
        const int col = (c & 3) * 8;
        async_b128(lds_addr(Vts) + (unsigned)c * 16u,
                   gvt + (size_t)d * L_ + col);
      }
      if (kt2 + 2 <= ktmax)  // prefetch next PR block (4KB = 64 lines)
        __builtin_prefetch(PR + (size_t)(krb + 32) * K_ + (tid & 63) * 32, 0, 0);
      wait_async0();
    }
    __syncthreads();

    if (kt2 <= qt) {  // wave-uniform: EXEC stays full for WMMA
      v8f s0 = {}, s1 = {};
      {
        v16h a0 = load_fragA(PRs, K_);
        v16h a1 = load_fragA(PRs + 32, K_);
        v16h c0 = load_fragA(PIMs, K_);
        v16h c1 = load_fragA(PIMs + 32, K_);
        s0 = wmma_f16(a0, prq0, s0);
        s0 = wmma_f16(a1, prq1, s0);
        s0 = wmma_f16(c0, piq0, s0);
        s0 = wmma_f16(c1, piq1, s0);
        if (kt2 == qt) {
#pragma unroll
          for (int j = 0; j < 8; ++j)
            if ((j + 8 * g) > r) s0[j] = 0.0f;
        }
      }
      if (kt2 + 1 <= qt) {
        v16h a0 = load_fragA(PRs + 16 * K_, K_);
        v16h a1 = load_fragA(PRs + 16 * K_ + 32, K_);
        v16h c0 = load_fragA(PIMs + 16 * K_, K_);
        v16h c1 = load_fragA(PIMs + 16 * K_ + 32, K_);
        s1 = wmma_f16(a0, prq0, s1);
        s1 = wmma_f16(a1, prq1, s1);
        s1 = wmma_f16(c0, piq0, s1);
        s1 = wmma_f16(c1, piq1, s1);
        if (kt2 + 1 == qt) {
#pragma unroll
          for (int j = 0; j < 8; ++j)
            if ((j + 8 * g) > r) s1[j] = 0.0f;
        }
      }
      // S A-fragment: elements 0..7 = s0 regs, 8..15 = s1 regs (no shuffles)
      v16h sa;
#pragma unroll
      for (int j = 0; j < 8; ++j) {
        sa[j]     = (_Float16)s0[j];
        sa[8 + j] = (_Float16)s1[j];
      }
      // retrieved += S(16x32) @ V(32 x 16) over all D column tiles
#pragma unroll
      for (int ct = 0; ct < D_ / 16; ++ct) {
        v16h bf = load_fragBt(Vts + (ct * 16) * 32, 32);
        acc[ct] = wmma_f16(sa, bf, acc[ct]);
      }
    }
    __syncthreads();
  }

  // store retrieved TRANSPOSED: RETt[d][row]; contiguous 16B f32 stores
#pragma unroll
  for (int ct = 0; ct < D_ / 16; ++ct) {
    const int d = ct * 16 + r;
    float* p = RETt + (size_t)d * N_ + rb + 8 * g;
    *(v4f*)p       = __builtin_shufflevector(acc[ct], acc[ct], 0, 1, 2, 3);
    *(v4f*)(p + 4) = __builtin_shufflevector(acc[ct], acc[ct], 4, 5, 6, 7);
  }
}

// ---------------------------------------------------------------------------
// Kernel 3: scale by 1/sqrt((l+1)*K) + LayerNorm -> f16 (row-major RNh)
// ---------------------------------------------------------------------------
__global__ __launch_bounds__(256) void ln_kernel(
    const float* __restrict__ RETt, const float* __restrict__ ln_g,
    const float* __restrict__ ln_b, _Float16* __restrict__ RNh) {
  __shared__ float red[256];
  const int row = blockIdx.x;
  const int tid = threadIdx.x;
  const int l = row & (L_ - 1);
  const float scale = rsqrtf((float)(l + 1) * (float)K_);
  const float v = RETt[(size_t)tid * N_ + row] * scale;

  red[tid] = v;
  __syncthreads();
  for (int s = 128; s > 0; s >>= 1) {
    if (tid < s) red[tid] += red[tid + s];
    __syncthreads();
  }
  const float mu = red[0] * (1.0f / D_);
  __syncthreads();
  const float dv = v - mu;
  red[tid] = dv * dv;
  __syncthreads();
  for (int s = 128; s > 0; s >>= 1) {
    if (tid < s) red[tid] += red[tid + s];
    __syncthreads();
  }
  const float var = red[0] * (1.0f / D_);
  RNh[(size_t)row * D_ + tid] =
      (_Float16)(dv * rsqrtf(var + 1e-5f) * ln_g[tid] + ln_b[tid]);
}

// ---------------------------------------------------------------------------
// Kernel 4: out = x + rn@Wo + bo (WMMA); result staged in LDS so the final
// residual-add writes are coalesced row-major b128 stores.
// ---------------------------------------------------------------------------
__global__ __launch_bounds__(256) void out_kernel(
    const _Float16* __restrict__ RNh, const _Float16* __restrict__ Wot,
    const float* __restrict__ bo, const float* __restrict__ x,
    float* __restrict__ out) {
  __shared__ __attribute__((aligned(16))) _Float16 rt[16 * D_];  // 8 KB
  __shared__ __attribute__((aligned(16))) float    ot[16 * D_];  // 16 KB
  const int row0 = blockIdx.x * 16;
  const int tid  = threadIdx.x;
  const int wave = tid >> 5;
  const int lane = tid & 31;
  const int g = lane >> 4, r = lane & 15;

  {
    const int m = tid >> 4, c0 = (tid & 15) * 16;
    const _Float16* grn = RNh + (size_t)(row0 + m) * D_ + c0;
    const unsigned l0 = lds_addr(rt) + (unsigned)(m * D_ + c0) * 2u;
    async_b128(l0, grn);
    async_b128(l0 + 16u, grn + 8);
    wait_async0();
  }
  __syncthreads();

  for (int ct = wave; ct < D_ / 16; ct += 8) {
    v8f acc = {};
#pragma unroll
    for (int ks = 0; ks < D_ / 32; ++ks) {
      v16h a = load_fragA(rt + ks * 32, D_);
      v16h b = load_fragBt(Wot + (ct * 16) * D_ + ks * 32, D_);
      acc = wmma_f16(a, b, acc);
    }
    const int n = ct * 16 + r;
#pragma unroll
    for (int j = 0; j < 8; ++j) ot[(j + 8 * g) * D_ + n] = acc[j];
  }
  __syncthreads();

  // coalesced epilogue: each thread owns 16 contiguous output floats
  {
    const int m = tid >> 4, c0 = (tid & 15) * 16;
    const size_t base = (size_t)(row0 + m) * D_ + c0;
#pragma unroll
    for (int u = 0; u < 16; u += 4) {
      v4f xv = *(const v4f*)(x + base + u);
      v4f ov;
#pragma unroll
      for (int w = 0; w < 4; ++w)
        ov[w] = xv[w] + ot[m * D_ + c0 + u + w] + bo[c0 + u + w];
      *(v4f*)(out + base + u) = ov;
    }
  }
}

// ---------------------------------------------------------------------------
// Host launcher
// ---------------------------------------------------------------------------
static inline size_t align256(size_t x) { return (x + 255) & ~(size_t)255; }

extern "C" void kernel_launch(void* const* d_in, const int* in_sizes, int n_in,
                              void* d_out, int out_size, void* d_ws,
                              size_t ws_size, hipStream_t stream) {
  const float* x   = (const float*)d_in[0];
  const float* pos = (const float*)d_in[1];
  const float* W1  = (const float*)d_in[2];
  const float* b1  = (const float*)d_in[3];
  const float* W2  = (const float*)d_in[4];
  const float* b2  = (const float*)d_in[5];
  const float* Wa  = (const float*)d_in[6];
  const float* ba  = (const float*)d_in[7];
  const float* Wv  = (const float*)d_in[8];
  const float* bv  = (const float*)d_in[9];
  const float* lng = (const float*)d_in[10];
  const float* lnb = (const float*)d_in[11];
  const float* Wo  = (const float*)d_in[12];
  const float* bo  = (const float*)d_in[13];
  float* out = (float*)d_out;

  char* p = (char*)d_ws;
  auto alloc = [&](size_t bytes) {
    char* q = p;
    p += align256(bytes);
    return q;
  };
  _Float16* xh  = (_Float16*)alloc((size_t)N_ * D_ * 2);
  _Float16* W1t = (_Float16*)alloc((size_t)D_ * D_ * 2);
  _Float16* W2t = (_Float16*)alloc((size_t)D_ * K_ * 2);
  _Float16* Wat = (_Float16*)alloc((size_t)D_ * K_ * 2);
  _Float16* Wvt = (_Float16*)alloc((size_t)D_ * D_ * 2);
  _Float16* Wot = (_Float16*)alloc((size_t)D_ * D_ * 2);
  _Float16* PR  = (_Float16*)alloc((size_t)N_ * K_ * 2);
  _Float16* PIM = (_Float16*)alloc((size_t)N_ * K_ * 2);
  _Float16* Vt  = (_Float16*)alloc((size_t)N_ * D_ * 2);
  _Float16* RNh = (_Float16*)alloc((size_t)N_ * D_ * 2);
  float*    RETt = (float*)  alloc((size_t)N_ * D_ * 4);

  cvt_f32_to_f16<<<(N_ * D_ + 255) / 256, 256, 0, stream>>>(x, xh, N_ * D_);
  // transposed f16 weights: Wt[out][in]
  cvt_transpose<<<(D_ * D_ + 255) / 256, 256, 0, stream>>>(W1, W1t, D_, D_);
  cvt_transpose<<<(D_ * K_ + 255) / 256, 256, 0, stream>>>(W2, W2t, D_, K_);
  cvt_transpose<<<(D_ * K_ + 255) / 256, 256, 0, stream>>>(Wa, Wat, D_, K_);
  cvt_transpose<<<(D_ * D_ + 255) / 256, 256, 0, stream>>>(Wv, Wvt, D_, D_);
  cvt_transpose<<<(D_ * D_ + 255) / 256, 256, 0, stream>>>(Wo, Wot, D_, D_);

  encoder_kernel<<<N_ / 16, 256, 0, stream>>>(xh, W1t, b1, W2t, b2, Wat, ba,
                                              Wvt, bv, pos, PR, PIM, Vt);
  attn_kernel<<<B_ * (L_ / 64), 128, 0, stream>>>(PR, PIM, Vt, RETt);
  ln_kernel<<<N_, 256, 0, stream>>>(RETt, lng, lnb, RNh);
  out_kernel<<<N_ / 16, 256, 0, stream>>>(RNh, Wot, bo, x, out);
}
